// GNN_24361054502958
// MI455X (gfx1250) — matrix operand order repaired and checked by reading, not compiled, not run on hardware
//
#include <hip/hip_runtime.h>

// ---------------------------------------------------------------- constants
constexpr int NN   = 40000;    // nodes
constexpr int NE   = 640000;   // edges
constexpr int DD   = 128;      // node/edge dim
constexpr int EMB  = 256;      // concat dim
constexpr int NL   = 3;        // layers
constexpr float BN_EPS = 1e-5f;

typedef __attribute__((ext_vector_type(16))) __bf16 v16bf;
typedef __attribute__((ext_vector_type(8)))  __bf16 v8bf;
typedef __attribute__((ext_vector_type(4)))  __bf16 v4bf;
typedef __attribute__((ext_vector_type(8)))  float  v8f;
typedef __attribute__((ext_vector_type(4)))  float  v4f;

#define BF_PAD 8    // bf16 row pad -> 528B stride, conflict-free ds_load_b128
#define F_PAD  4    // f32 row pad  -> 1040B stride, conflict-free

// ------------------------------------------------ CDNA5 async copy helpers
#if __has_builtin(__builtin_amdgcn_global_load_async_to_lds_b128)
#define HAVE_ASYNC_BUILTIN 1
#else
#define HAVE_ASYNC_BUILTIN 0
#endif
#if __has_builtin(__builtin_amdgcn_s_wait_asynccnt)
#define HAVE_WAITASYNC_BUILTIN 1
#else
#define HAVE_WAITASYNC_BUILTIN 0
#endif

// builtin expects pointers to <4 x i32> (b128 payload) in global/LDS addr spaces
typedef int v4i_vs __attribute__((vector_size(16)));
typedef __attribute__((address_space(1))) v4i_vs gv4i_t;
typedef __attribute__((address_space(3))) v4i_vs lv4i_t;
typedef __attribute__((address_space(3))) void   lds_void;

__device__ __forceinline__ void async_b128(void* lds, const void* g) {
#if HAVE_ASYNC_BUILTIN
    __builtin_amdgcn_global_load_async_to_lds_b128(
        (gv4i_t*)g, (lv4i_t*)lds, 0, 0);
#else
    asm volatile("global_load_async_to_lds_b128 %0, %1, off"
                 :: "v"((unsigned int)(uintptr_t)(lds_void*)lds),
                    "v"((unsigned long long)(uintptr_t)g)
                 : "memory");
#endif
}

template <int N>
__device__ __forceinline__ void wait_async() {
#if HAVE_WAITASYNC_BUILTIN
    __builtin_amdgcn_s_wait_asynccnt((short)N);
#else
    asm volatile("s_wait_asynccnt %0" :: "i"(N) : "memory");
#endif
}

__device__ __forceinline__ void atomAddF(float* p, float v) {
    unsafeAtomicAdd(p, v);   // ds_add_f32 / global_atomic_add_f32
}

__device__ __forceinline__ v16bf cat16(v8bf lo, v8bf hi) {
    return __builtin_shufflevector(lo, hi, 0,1,2,3,4,5,6,7,8,9,10,11,12,13,14,15);
}

// ----------------------------------------------------- weight fp32 -> bf16
__global__ __launch_bounds__(256)
void gin_cvt_w(const float* __restrict__ W1, const float* __restrict__ W2,
               __bf16* __restrict__ W1bf, __bf16* __restrict__ W2bf) {
    const int n1 = NL * EMB * EMB;
    const int n2 = NL * DD * EMB;
    int i = blockIdx.x * blockDim.x + threadIdx.x;
    if (i < n1)            W1bf[i]      = (__bf16)W1[i];
    else if (i < n1 + n2)  W2bf[i - n1] = (__bf16)W2[i - n1];
}

// --------------------------------------------------------------- pass A
// x = concat(h[src], e) @ W1^T + b1 ; store x(bf16) ; accumulate col sum/sumsq
// Async double-buffered f32 staging -> fixup (relu + cvt bf16) -> WMMA.
template <int RELU>
__global__ __launch_bounds__(256)
void gin_gemm1(const float* __restrict__ h_in, const float* __restrict__ e_in,
               const __bf16* __restrict__ W1bf, const float* __restrict__ b1,
               const int* __restrict__ src, __bf16* __restrict__ x_out,
               float* __restrict__ stat_sum, float* __restrict__ stat_sq)
{
    __shared__ __align__(16) float  Fbuf[2][32][EMB + F_PAD];   // raw f32 tiles
    __shared__ __align__(32) __bf16 Ash[32][EMB + BF_PAD];      // converted tile
    __shared__ float sSum[EMB];
    __shared__ float sSq[EMB];

    const int tid  = threadIdx.x;
    const int wave = tid >> 5;
    const int lane = tid & 31;
    const int mtile = wave & 1;          // 2 M-tiles of 16 rows
    const int nt0   = (wave >> 1) * 4;   // 4 N-tiles per wave (of 16)
    const int lm    = lane & 15;
    const int lhi   = lane >> 4;
    const int kb    = lhi * 8;

    for (int c = tid; c < EMB; c += 256) { sSum[c] = 0.f; sSq[c] = 0.f; }

    const int ntiles = NE / 32;
    const int stride = gridDim.x;

    // issue async stage of first tile (8 x b128 per thread)
    {
        const int r0 = blockIdx.x * 32;
        for (int idx = tid; idx < 32 * 64; idx += 256) {
            const int row = idx >> 6;
            const int q   = idx & 63;
            const int r   = r0 + row;
            const float* g = (q < 32) ? (h_in + (size_t)src[r] * DD + q * 4)
                                      : (e_in + (size_t)r * DD + (q - 32) * 4);
            async_b128(&Fbuf[0][row][q * 4], g);
        }
    }

    int b = 0;
    for (int t = blockIdx.x; t < ntiles; t += stride) {
        const int r0 = t * 32;
        const int tn = t + stride;
        if (tn < ntiles) {              // prefetch next tile into other buffer
            const int rn = tn * 32;
            for (int idx = tid; idx < 32 * 64; idx += 256) {
                const int row = idx >> 6;
                const int q   = idx & 63;
                const int r   = rn + row;
                const float* g = (q < 32) ? (h_in + (size_t)src[r] * DD + q * 4)
                                          : (e_in + (size_t)r * DD + (q - 32) * 4);
                async_b128(&Fbuf[b ^ 1][row][q * 4], g);
            }
            wait_async<8>();            // oldest 8 (current tile) complete
        } else {
            wait_async<0>();
        }
        __syncthreads();

        // fixup: f32 -> (relu) -> bf16 compact tile
        for (int idx = tid; idx < 32 * 64; idx += 256) {
            const int row = idx >> 6;
            const int q   = idx & 63;
            v4f v = *(const v4f*)(&Fbuf[b][row][q * 4]);
            if (RELU) {
                v[0] = fmaxf(v[0], 0.f); v[1] = fmaxf(v[1], 0.f);
                v[2] = fmaxf(v[2], 0.f); v[3] = fmaxf(v[3], 0.f);
            }
            v4bf o;
            o[0] = (__bf16)v[0]; o[1] = (__bf16)v[1];
            o[2] = (__bf16)v[2]; o[3] = (__bf16)v[3];
            *(v4bf*)(&Ash[row][q * 4]) = o;
        }
        __syncthreads();

        v8f acc[4] = {};
        const int arow = mtile * 16 + lm;
        #pragma unroll
        for (int ks = 0; ks < 8; ++ks) {
            const int k0 = ks * 32;
            v8bf alo = *(const v8bf*)(&Ash[arow][k0 + kb]);
            v8bf ahi = *(const v8bf*)(&Ash[arow][k0 + 16 + kb]);
            v16bf a = cat16(alo, ahi);
            #pragma unroll
            for (int i = 0; i < 4; ++i) {
                const int n = (nt0 + i) * 16 + lm;
                v16bf bw = cat16(*(const v8bf*)(W1bf + (size_t)n * EMB + k0 + kb), *(const v8bf*)(W1bf + (size_t)n * EMB + k0 + 16 + kb));
                acc[i] = __builtin_amdgcn_wmma_f32_16x16x32_bf16(
                    false, a, false, bw, (short)0, acc[i], false, false);
            }
        }

        #pragma unroll
        for (int i = 0; i < 4; ++i) {
            const int n = (nt0 + i) * 16 + lm;
            const float bias = b1[n];
            float lsum = 0.f, lsq = 0.f;
            #pragma unroll
            for (int j = 0; j < 8; ++j) {
                float v = acc[i][j] + bias;
                lsum += v; lsq += v * v;
                const int rowl = mtile * 16 + lhi * 8 + j;
                x_out[(size_t)(r0 + rowl) * EMB + n] = (__bf16)v;
            }
            atomAddF(&sSum[n], lsum);
            atomAddF(&sSq[n],  lsq);
        }
        __syncthreads();    // all reads of Fbuf[b]/Ash done before reuse
        b ^= 1;
    }
    __syncthreads();
    for (int c = tid; c < EMB; c += 256) {
        atomAddF(&stat_sum[c], sSum[c]);
        atomAddF(&stat_sq[c],  sSq[c]);
    }
}

// ----------------------------------------------- BN stats -> affine coeffs
__global__ void gin_bn_finalize(const float* __restrict__ sum, const float* __restrict__ sq,
                                const float* __restrict__ gamma, const float* __restrict__ beta,
                                float* __restrict__ scale, float* __restrict__ shift) {
    const int c = threadIdx.x;                 // 256 threads
    const float inv = 1.0f / (float)NE;
    const float mu  = sum[c] * inv;
    const float var = sq[c] * inv - mu * mu;
    const float g   = gamma[c] * rsqrtf(var + BN_EPS);
    scale[c] = g;
    shift[c] = beta[c] - mu * g;
}

// --------------------------------------------------------------- pass B
// y = relu(x*scale+shift) ; m = y @ W2^T + b2 ; atomic scatter to h_out[dst]
// Async double-buffered bf16 staging -> fixup (affine+relu) -> WMMA.
__global__ __launch_bounds__(256)
void gin_gemm2(const __bf16* __restrict__ x_in, const __bf16* __restrict__ W2bf,
               const float* __restrict__ b2, const float* __restrict__ bn_scale,
               const float* __restrict__ bn_shift, const int* __restrict__ dst,
               float* __restrict__ h_out)
{
    __shared__ __align__(16) __bf16 Xbuf[2][32][EMB + BF_PAD];  // raw x tiles
    __shared__ __align__(32) __bf16 Ysh[32][EMB + BF_PAD];      // normalized
    __shared__ float sScale[EMB];
    __shared__ float sShift[EMB];

    const int tid  = threadIdx.x;
    const int wave = tid >> 5;
    const int lane = tid & 31;
    const int mtile = wave & 1;
    const int nt0   = (wave >> 1) * 2;   // 2 N-tiles per wave, N total = 128
    const int lm    = lane & 15;
    const int lhi   = lane >> 4;
    const int kb    = lhi * 8;

    for (int c = tid; c < EMB; c += 256) { sScale[c] = bn_scale[c]; sShift[c] = bn_shift[c]; }

    const int ntiles = NE / 32;
    const int stride = gridDim.x;

    {   // first tile: 4 x b128 per thread
        const int r0 = blockIdx.x * 32;
        for (int idx = tid; idx < 32 * 32; idx += 256) {
            const int row = idx >> 5;
            const int q   = idx & 31;
            async_b128(&Xbuf[0][row][q * 8],
                       x_in + (size_t)(r0 + row) * EMB + q * 8);
        }
    }

    int b = 0;
    for (int t = blockIdx.x; t < ntiles; t += stride) {
        const int r0 = t * 32;
        const int tn = t + stride;
        if (tn < ntiles) {
            const int rn = tn * 32;
            for (int idx = tid; idx < 32 * 32; idx += 256) {
                const int row = idx >> 5;
                const int q   = idx & 31;
                async_b128(&Xbuf[b ^ 1][row][q * 8],
                           x_in + (size_t)(rn + row) * EMB + q * 8);
            }
            wait_async<4>();
        } else {
            wait_async<0>();
        }
        __syncthreads();

        // fixup: BN affine + relu, bf16 -> bf16
        for (int idx = tid; idx < 32 * 32; idx += 256) {
            const int row = idx >> 5;
            const int q   = idx & 31;
            v8bf xv = *(const v8bf*)(&Xbuf[b][row][q * 8]);
            v8bf o;
            #pragma unroll
            for (int i = 0; i < 8; ++i) {
                const int c = q * 8 + i;
                float f = (float)xv[i];
                f = fmaxf(f * sScale[c] + sShift[c], 0.f);
                o[i] = (__bf16)f;
            }
            *(v8bf*)(&Ysh[row][q * 8]) = o;
        }
        __syncthreads();

        v8f acc[2] = {};
        const int arow = mtile * 16 + lm;
        #pragma unroll
        for (int ks = 0; ks < 8; ++ks) {
            const int k0 = ks * 32;
            v8bf alo = *(const v8bf*)(&Ysh[arow][k0 + kb]);
            v8bf ahi = *(const v8bf*)(&Ysh[arow][k0 + 16 + kb]);
            v16bf a = cat16(alo, ahi);
            #pragma unroll
            for (int i = 0; i < 2; ++i) {
                const int n = (nt0 + i) * 16 + lm;
                v16bf bw = cat16(*(const v8bf*)(W2bf + (size_t)n * EMB + k0 + kb), *(const v8bf*)(W2bf + (size_t)n * EMB + k0 + 16 + kb));
                acc[i] = __builtin_amdgcn_wmma_f32_16x16x32_bf16(
                    false, a, false, bw, (short)0, acc[i], false, false);
            }
        }

        #pragma unroll
        for (int i = 0; i < 2; ++i) {
            const int n = (nt0 + i) * 16 + lm;
            const float bias = b2[n];
            #pragma unroll
            for (int j = 0; j < 8; ++j) {
                const int rowl = mtile * 16 + lhi * 8 + j;
                const int dn = dst[r0 + rowl];
                atomAddF(&h_out[(size_t)dn * DD + n], acc[i][j] + bias);
            }
        }
        __syncthreads();
        b ^= 1;
    }
}

// ------------------------------------------------------------------ launch
extern "C" void kernel_launch(void* const* d_in, const int* in_sizes, int n_in,
                              void* d_out, int out_size, void* d_ws, size_t ws_size,
                              hipStream_t stream) {
    const float* h     = (const float*)d_in[0];
    const float* e     = (const float*)d_in[1];
    const float* W1    = (const float*)d_in[2];
    const float* b1    = (const float*)d_in[3];
    const float* gamma = (const float*)d_in[4];
    const float* beta  = (const float*)d_in[5];
    const float* W2    = (const float*)d_in[6];
    const float* b2    = (const float*)d_in[7];
    const int*   src   = (const int*)d_in[8];
    const int*   dst   = (const int*)d_in[9];
    float* hout = (float*)d_out;

    char* ws = (char*)d_ws;
    size_t off = 0;
    __bf16* xbf  = (__bf16*)(ws + off); off += (size_t)NE * EMB * 2;     // 327.7 MB
    float*  hA   = (float*)(ws + off);  off += (size_t)NN * DD * 4;      // 20.5 MB
    float*  hB   = (float*)(ws + off);  off += (size_t)NN * DD * 4;      // 20.5 MB
    __bf16* W1bf = (__bf16*)(ws + off); off += (size_t)NL * EMB * EMB * 2;
    __bf16* W2bf = (__bf16*)(ws + off); off += (size_t)NL * DD * EMB * 2;
    float*  ssum = (float*)(ws + off);  off += EMB * sizeof(float);
    float*  ssq  = (float*)(ws + off);  off += EMB * sizeof(float);
    float*  bsc  = (float*)(ws + off);  off += EMB * sizeof(float);
    float*  bsh  = (float*)(ws + off);  off += EMB * sizeof(float);

    {
        const int nw = NL * EMB * EMB + NL * DD * EMB;
        gin_cvt_w<<<(nw + 255) / 256, 256, 0, stream>>>(W1, W2, W1bf, W2bf);
    }

    const int GBLK = 2000;   // grid-stride over 20000 32-row tiles
    for (int l = 0; l < NL; ++l) {
        const float* hin   = (l == 0) ? h  : ((l == 1) ? hA : hB);
        float*       hnext = (l == 0) ? hA : ((l == 1) ? hB : hout);
        const __bf16* w1l = W1bf + (size_t)l * EMB * EMB;
        const __bf16* w2l = W2bf + (size_t)l * DD * EMB;

        (void)hipMemsetAsync(ssum, 0, 2 * EMB * sizeof(float), stream);  // ssum+ssq
        if (l == 0)
            gin_gemm1<0><<<GBLK, 256, 0, stream>>>(hin, e, w1l, b1 + l * EMB, src,
                                                   xbf, ssum, ssq);
        else
            gin_gemm1<1><<<GBLK, 256, 0, stream>>>(hin, e, w1l, b1 + l * EMB, src,
                                                   xbf, ssum, ssq);
        gin_bn_finalize<<<1, 256, 0, stream>>>(ssum, ssq, gamma + l * EMB,
                                               beta + l * EMB, bsc, bsh);
        (void)hipMemsetAsync(hnext, 0, (size_t)NN * DD * sizeof(float), stream);
        gin_gemm2<<<GBLK, 256, 0, stream>>>(xbf, w2l, b2 + l * DD, bsc, bsh, dst, hnext);
    }
    (void)in_sizes; (void)n_in; (void)out_size; (void)ws_size;
}
